// SelfCorrelationComputation2_584115552554
// MI455X (gfx1250) — matrix operand order repaired and checked by reading, not compiled, not run on hardware
//
#include <hip/hip_runtime.h>
#include <stddef.h>

// ---------------- problem constants ----------------
#define B_   8
#define C_   128
#define H_   48
#define W_   48
#define K_   7
#define PAD_ 3
#define HW_  (H_ * W_)          // 2304
#define HP_  (H_ + 2 * PAD_)    // 54
#define WP_  (W_ + 2 * PAD_)    // 54
#define HWP_ (HP_ * WP_)        // 2916
#define NPIX_ (B_ * C_ * HW_)   // 2359296
#define EPS_ 1e-5f
#define BNCOUNT_ ((float)(B_ * HW_))  // 18432

typedef __attribute__((ext_vector_type(16))) __bf16 v16bf;
typedef __attribute__((ext_vector_type(8)))  float  v8f;

union FragU { v16bf v; uint4 q[2]; };

// fp32 -> bf16, round-to-nearest-even (inputs are finite gaussians; skip NaN path)
static __device__ __forceinline__ unsigned short f2bf(float f) {
  unsigned int u = __float_as_uint(f);
  unsigned int r = u + 0x7FFFu + ((u >> 16) & 1u);
  return (unsigned short)(r >> 16);
}

// A fragment (16-bit A, 16x32): lane m=lane&15, half=lane>>4.
// element j: K = j + (j>=8 ? 8 : 0) + half*8   -> two contiguous 8-elem chunks.
static __device__ __forceinline__ v16bf load_afrag(const unsigned short* row_k0, int half) {
  FragU f;
  const unsigned short* p = row_k0 + half * 8;
  f.q[0] = *(const uint4*)(p);
  f.q[1] = *(const uint4*)(p + 16);
  return f.v;
}

// B fragment (16-bit B, 32x16): lane n=lane&15, half=lane>>4.
// element j: K = j + (j>=8 ? 8 : 0) + half*8   -> two contiguous 8-elem chunks (same as A).
static __device__ __forceinline__ v16bf load_bfrag(const unsigned short* row_k0, int half) {
  FragU f;
  const unsigned short* p = row_k0 + half * 8;
  f.q[0] = *(const uint4*)(p);
  f.q[1] = *(const uint4*)(p + 16);
  return f.v;
}

// ---------------- utility kernels ----------------
__global__ void zero_f(float* p, int n) {
  int i = blockIdx.x * blockDim.x + threadIdx.x;
  if (i < n) p[i] = 0.0f;
}

// pack Wq | Wk | Wv | W_agg (each 128x128 row-major [o][c]) into bf16
__global__ void conv_w(const float* __restrict__ Wq, const float* __restrict__ Wk,
                       const float* __restrict__ Wv, const float* __restrict__ Wa,
                       unsigned short* __restrict__ out) {
  int i = blockIdx.x * 256 + threadIdx.x;
  if (i >= 4 * C_ * C_) return;
  int which = i >> 14;           // /16384
  int j = i & 16383;
  const float* src = (which == 0) ? Wq : (which == 1) ? Wk : (which == 2) ? Wv : Wa;
  out[i] = f2bf(src[j]);
}

// x [B][C][HW] f32 -> xT [B][HW][C] bf16 (GEMM B-operand, K contiguous)
__global__ void transpose_x(const float* __restrict__ x, unsigned short* __restrict__ xT) {
  int i = blockIdx.x * 256 + threadIdx.x;   // exact: NPIX_ / 256 blocks
  int hw = i % HW_;
  int bc = i / HW_;
  int c = bc & (C_ - 1);
  int b = bc / C_;
  xT[((size_t)b * HW_ + hw) * C_ + c] = f2bf(x[i]);
}

// ---------------- WMMA GEMM: Q/K/V (1x1 convs) ----------------
// One wave computes a 16(M) x 64(N) tile: 4 f32 accumulators sharing one A
// fragment per K-step (weights fetched once per 64 columns -> 4x less A traffic,
// 16 WMMAs per wave). grid (36, 8, B*3), block 32.
// s=0 -> q[b][m][hw]; s=1/2 -> k/v scattered into padded buffers (halo pre-zeroed).
__global__ __launch_bounds__(32) void gemm_qkv(const unsigned short* __restrict__ Wall,
                                               const unsigned short* __restrict__ xT,
                                               float* __restrict__ qbuf,
                                               float* __restrict__ kpad,
                                               float* __restrict__ vpad) {
  const int lane = threadIdx.x;
  const int nt = blockIdx.x, mt = blockIdx.y, z = blockIdx.z;
  const int b = z / 3, s = z % 3;
  const int m = lane & 15, half = lane >> 4;
  const int n0 = nt * 64;

  const unsigned short* arow = Wall + (size_t)s * C_ * C_ + (size_t)(mt * 16 + m) * C_;
  const unsigned short* brow0 = xT + ((size_t)b * HW_ + n0 + m) * C_;

  v8f acc0 = {}, acc1 = {}, acc2 = {}, acc3 = {};
#pragma unroll
  for (int k0 = 0; k0 < C_; k0 += 32) {
    v16bf a  = load_afrag(arow + k0, half);
    v16bf b0 = load_bfrag(brow0 + k0, half);
    v16bf b1 = load_bfrag(brow0 + 16 * C_ + k0, half);
    v16bf b2 = load_bfrag(brow0 + 32 * C_ + k0, half);
    v16bf b3 = load_bfrag(brow0 + 48 * C_ + k0, half);
    acc0 = __builtin_amdgcn_wmma_f32_16x16x32_bf16(false, a, false, b0, (short)0, acc0, false, false);
    acc1 = __builtin_amdgcn_wmma_f32_16x16x32_bf16(false, a, false, b1, (short)0, acc1, false, false);
    acc2 = __builtin_amdgcn_wmma_f32_16x16x32_bf16(false, a, false, b2, (short)0, acc2, false, false);
    acc3 = __builtin_amdgcn_wmma_f32_16x16x32_bf16(false, a, false, b3, (short)0, acc3, false, false);
  }

  v8f accs[4] = {acc0, acc1, acc2, acc3};
  if (s == 0) {
    float* dst = qbuf + ((size_t)(b * C_ + mt * 16)) * HW_;
#pragma unroll
    for (int j = 0; j < 4; ++j) {
      const int n = n0 + j * 16 + m;     // this lane's N column for subtile j
#pragma unroll
      for (int r = 0; r < 8; ++r) {
        int mm = r + 8 * half;           // D layout: VGPR r -> M=r (lanes<16) / r+8
        dst[(size_t)mm * HW_ + n] = accs[j][r];
      }
    }
  } else {
    float* pb = (s == 1) ? kpad : vpad;
    size_t cb = ((size_t)(b * C_ + mt * 16)) * HWP_;
#pragma unroll
    for (int j = 0; j < 4; ++j) {
      const int n = n0 + j * 16 + m;
      const int h = n / W_, w = n % W_;
      size_t base = cb + (size_t)(h + PAD_) * WP_ + (w + PAD_);
#pragma unroll
      for (int r = 0; r < 8; ++r) {
        int mm = r + 8 * half;
        pb[base + (size_t)mm * HWP_] = accs[j][r];
      }
    }
  }
}

// ---------------- attention: per-channel 7x7 softmax ----------------
// grid (9, C, B), block 256. Reads q in place, writes attention output over it.
// Also accumulates BN1 per-channel sum / sum-of-squares via LDS tree + 1 atomic.
__global__ __launch_bounds__(256) void attn_kernel(float* __restrict__ qatt,
                                                   const float* __restrict__ kpad,
                                                   const float* __restrict__ vpad,
                                                   const float* __restrict__ rel_h,
                                                   const float* __restrict__ rel_w,
                                                   float* __restrict__ sum1,
                                                   float* __restrict__ sumsq1) {
  const int tid = threadIdx.x;
  const int hw = blockIdx.x * 256 + tid;  // exact coverage of 2304
  const int c = blockIdx.y, b = blockIdx.z;
  const int h = hw / W_, w = hw % W_;

  const size_t qi = ((size_t)(b * C_ + c)) * HW_ + hw;
  const float q = qatt[qi];
  const float* kb = kpad + ((size_t)(b * C_ + c)) * HWP_ + (size_t)h * WP_ + w;
  const float* vb = vpad + ((size_t)(b * C_ + c)) * HWP_ + (size_t)h * WP_ + w;

  const bool useH = (c < C_ / 2);
  const float* rel = useH ? (rel_h + c * K_) : (rel_w + (c - C_ / 2) * K_);
  float bias7[K_];
#pragma unroll
  for (int i = 0; i < K_; ++i) bias7[i] = rel[i];

  float logit[K_ * K_];
  float mx = -3.4e38f;
#pragma unroll
  for (int kh = 0; kh < K_; ++kh) {
#pragma unroll
    for (int kw = 0; kw < K_; ++kw) {
      float kv = kb[kh * WP_ + kw];
      float l = q * (kv + (useH ? bias7[kh] : bias7[kw]));
      logit[kh * K_ + kw] = l;
      mx = fmaxf(mx, l);
    }
  }

  float den = 0.0f, acc = 0.0f;
#pragma unroll
  for (int t = 0; t < K_ * K_; ++t) {
    float e = __expf(logit[t] - mx);
    den += e;
    acc += e * vb[(t / K_) * WP_ + (t % K_)];
  }
  const float out = acc / den;
  qatt[qi] = out;                         // in-place: each thread owns its element

  __shared__ float s1[256], s2[256];
  s1[tid] = out;
  s2[tid] = out * out;
  __syncthreads();
  for (int off = 128; off > 0; off >>= 1) {
    if (tid < off) { s1[tid] += s1[tid + off]; s2[tid] += s2[tid + off]; }
    __syncthreads();
  }
  if (tid == 0) {
    atomicAdd(&sum1[c], s1[0]);
    atomicAdd(&sumsq1[c], s2[0]);
  }
}

// ---------------- BN parameter fold (train-mode batch stats) ----------------
__global__ void bn_params(const float* __restrict__ sum, const float* __restrict__ sumsq,
                          const float* __restrict__ gamma, const float* __restrict__ beta,
                          float* __restrict__ scale, float* __restrict__ shift) {
  int c = threadIdx.x;
  const float inv_n = 1.0f / BNCOUNT_;
  float mean = sum[c] * inv_n;
  float var = sumsq[c] * inv_n - mean * mean;   // population variance (ddof=0)
  float sc = gamma[c] * rsqrtf(var + EPS_);
  scale[c] = sc;
  shift[c] = beta[c] - mean * sc;
}

// y = relu(bn1(att)) -> bf16, transposed [B][HW][C] for GEMM2 B-operand
__global__ void make_y(const float* __restrict__ att, const float* __restrict__ scale,
                       const float* __restrict__ shift, unsigned short* __restrict__ ybf) {
  int i = blockIdx.x * 256 + threadIdx.x;
  int hw = i % HW_;
  int bc = i / HW_;
  int c = bc & (C_ - 1);
  int b = bc / C_;
  float v = att[i] * scale[c] + shift[c];
  v = fmaxf(v, 0.0f);
  ybf[((size_t)b * HW_ + hw) * C_ + c] = f2bf(v);
}

// ---------------- WMMA GEMM: aggregation 1x1 conv ----------------
// Same 16x64-per-wave tiling as gemm_qkv. grid (36, 8, B), block 32.
__global__ __launch_bounds__(32) void gemm_agg(const unsigned short* __restrict__ Wagg,
                                               const unsigned short* __restrict__ ybf,
                                               float* __restrict__ zbuf) {
  const int lane = threadIdx.x;
  const int nt = blockIdx.x, mt = blockIdx.y, b = blockIdx.z;
  const int m = lane & 15, half = lane >> 4;
  const int n0 = nt * 64;

  const unsigned short* arow = Wagg + (size_t)(mt * 16 + m) * C_;
  const unsigned short* brow0 = ybf + ((size_t)b * HW_ + n0 + m) * C_;

  v8f acc0 = {}, acc1 = {}, acc2 = {}, acc3 = {};
#pragma unroll
  for (int k0 = 0; k0 < C_; k0 += 32) {
    v16bf a  = load_afrag(arow + k0, half);
    v16bf b0 = load_bfrag(brow0 + k0, half);
    v16bf b1 = load_bfrag(brow0 + 16 * C_ + k0, half);
    v16bf b2 = load_bfrag(brow0 + 32 * C_ + k0, half);
    v16bf b3 = load_bfrag(brow0 + 48 * C_ + k0, half);
    acc0 = __builtin_amdgcn_wmma_f32_16x16x32_bf16(false, a, false, b0, (short)0, acc0, false, false);
    acc1 = __builtin_amdgcn_wmma_f32_16x16x32_bf16(false, a, false, b1, (short)0, acc1, false, false);
    acc2 = __builtin_amdgcn_wmma_f32_16x16x32_bf16(false, a, false, b2, (short)0, acc2, false, false);
    acc3 = __builtin_amdgcn_wmma_f32_16x16x32_bf16(false, a, false, b3, (short)0, acc3, false, false);
  }

  v8f accs[4] = {acc0, acc1, acc2, acc3};
  float* dst = zbuf + ((size_t)(b * C_ + mt * 16)) * HW_;
#pragma unroll
  for (int j = 0; j < 4; ++j) {
    const int n = n0 + j * 16 + m;
#pragma unroll
    for (int r = 0; r < 8; ++r) {
      int mm = r + 8 * half;
      dst[(size_t)mm * HW_ + n] = accs[j][r];
    }
  }
}

// per-channel sum / sumsq of z (for BN2)
__global__ __launch_bounds__(256) void chan_stats(const float* __restrict__ x,
                                                  float* __restrict__ sum,
                                                  float* __restrict__ sumsq) {
  const int tid = threadIdx.x;
  const int hw = blockIdx.x * 256 + tid;
  const int c = blockIdx.y, b = blockIdx.z;
  float v = x[((size_t)(b * C_ + c)) * HW_ + hw];
  __shared__ float s1[256], s2[256];
  s1[tid] = v;
  s2[tid] = v * v;
  __syncthreads();
  for (int off = 128; off > 0; off >>= 1) {
    if (tid < off) { s1[tid] += s1[tid + off]; s2[tid] += s2[tid + off]; }
    __syncthreads();
  }
  if (tid == 0) {
    atomicAdd(&sum[c], s1[0]);
    atomicAdd(&sumsq[c], s2[0]);
  }
}

__global__ void apply_bn2(const float* __restrict__ z, const float* __restrict__ scale,
                          const float* __restrict__ shift, float* __restrict__ out) {
  int i = blockIdx.x * 256 + threadIdx.x;
  int c = (i / HW_) & (C_ - 1);
  out[i] = z[i] * scale[c] + shift[c];
}

// ---------------- host-side launch ----------------
extern "C" void kernel_launch(void* const* d_in, const int* in_sizes, int n_in,
                              void* d_out, int out_size, void* d_ws, size_t ws_size,
                              hipStream_t stream) {
  (void)in_sizes; (void)n_in; (void)out_size; (void)ws_size;

  const float* x       = (const float*)d_in[0];
  const float* Wq      = (const float*)d_in[1];
  const float* Wk      = (const float*)d_in[2];
  const float* Wv      = (const float*)d_in[3];
  const float* rel_h   = (const float*)d_in[4];
  const float* rel_w   = (const float*)d_in[5];
  const float* W_agg   = (const float*)d_in[6];
  const float* bn1_g   = (const float*)d_in[7];
  const float* bn1_b   = (const float*)d_in[8];
  const float* bn2_g   = (const float*)d_in[9];
  const float* bn2_b   = (const float*)d_in[10];
  float* out = (float*)d_out;

  // workspace layout (256B aligned slices; aliasing is lifetime-safe)
  char* ws = (char*)d_ws;
  const size_t OFF_WBF  = 0;                                     // 4*128*128 u16
  const size_t OFF_XT   = OFF_WBF + (size_t)4 * C_ * C_ * 2;     // B*HW*C u16 (reused as ybf)
  const size_t OFF_Q    = OFF_XT + (size_t)B_ * HW_ * C_ * 2;    // B*C*HW f32 (reused as att)
  const size_t OFF_KPAD = OFF_Q + (size_t)NPIX_ * 4;             // B*C*HWP f32 (reused as z)
  const size_t OFF_VPAD = OFF_KPAD + (size_t)B_ * C_ * HWP_ * 4; // B*C*HWP f32
  const size_t OFF_ST   = OFF_VPAD + (size_t)B_ * C_ * HWP_ * 4; // 8*128 f32

  unsigned short* Wbf = (unsigned short*)(ws + OFF_WBF);
  unsigned short* xT  = (unsigned short*)(ws + OFF_XT);
  unsigned short* ybf = xT;                                  // alias: xT dead after gemm_qkv
  float* qatt = (float*)(ws + OFF_Q);                        // q, then attention output
  float* kpad = (float*)(ws + OFF_KPAD);
  float* vpad = (float*)(ws + OFF_VPAD);
  float* zbuf = kpad;                                        // alias: kpad dead after attn
  float* ST   = (float*)(ws + OFF_ST);
  float *sum1 = ST, *sq1 = ST + 128, *scale1 = ST + 256, *shift1 = ST + 384;
  float *sum2 = ST + 512, *sq2 = ST + 640, *scale2 = ST + 768, *shift2 = ST + 896;

  // 1) re-zero padded buffers + stat accumulators (graph-replay deterministic)
  {
    int n = 2 * B_ * C_ * HWP_;          // kpad+vpad contiguous
    zero_f<<<(n + 255) / 256, 256, 0, stream>>>(kpad, n);
    zero_f<<<4, 256, 0, stream>>>(ST, 1024);
  }
  // 2) weights + activations to bf16
  conv_w<<<(4 * C_ * C_ + 255) / 256, 256, 0, stream>>>(Wq, Wk, Wv, W_agg, Wbf);
  transpose_x<<<NPIX_ / 256, 256, 0, stream>>>(x, xT);
  // 3) Q/K/V GEMMs (WMMA, 16x64 tiles/wave), K/V scattered into padded buffers
  gemm_qkv<<<dim3(HW_ / 64, C_ / 16, B_ * 3), 32, 0, stream>>>(Wbf, xT, qatt, kpad, vpad);
  // 4) windowed softmax attention (in-place over q) + BN1 stats
  attn_kernel<<<dim3(HW_ / 256, C_, B_), 256, 0, stream>>>(qatt, kpad, vpad, rel_h, rel_w,
                                                           sum1, sq1);
  // 5) fold BN1, build y = relu(bn1(att)) in bf16 transposed
  bn_params<<<1, 128, 0, stream>>>(sum1, sq1, bn1_g, bn1_b, scale1, shift1);
  make_y<<<NPIX_ / 256, 256, 0, stream>>>(qatt, scale1, shift1, ybf);
  // 6) aggregation GEMM (WMMA, 16x64 tiles/wave)
  gemm_agg<<<dim3(HW_ / 64, C_ / 16, B_), 32, 0, stream>>>(Wbf + (size_t)3 * C_ * C_, ybf, zbuf);
  // 7) BN2 stats + fold + apply
  chan_stats<<<dim3(HW_ / 256, C_, B_), 256, 0, stream>>>(zbuf, sum2, sq2);
  bn_params<<<1, 128, 0, stream>>>(sum2, sq2, bn2_g, bn2_b, scale2, shift2);
  apply_bn2<<<NPIX_ / 256, 256, 0, stream>>>(zbuf, scale2, shift2, out);
}
